// GATEncoder_77695958385351
// MI455X (gfx1250) — compile-verified, matrix operand order checked by reading, not run on hardware
//
#include <hip/hip_runtime.h>
#include <hip/hip_bf16.h>

typedef __attribute__((ext_vector_type(16))) _Float16 v16h;
typedef __attribute__((ext_vector_type(8)))  _Float16 v8h;
typedef __attribute__((ext_vector_type(8)))  float    v8f;
typedef __attribute__((ext_vector_type(4)))  float    v4f;
typedef __attribute__((ext_vector_type(4)))  int      v4i;

// Problem constants
#define GB   8      // batch
#define GN   1024   // nodes
#define GFIN 256
#define GFOUT 512
#define GH   8      // heads
#define GD   64     // dims per head
#define GALPHA 0.2f

// ---------------------------------------------------------------------------
// k0: Wt[n][k] = (f16) W[k][n]   and zero s1|s2 (contiguous 131072 floats)
// ---------------------------------------------------------------------------
__global__ __launch_bounds__(256) void k0_prep(const float* __restrict__ W,
                                               _Float16* __restrict__ Wt,
                                               float* __restrict__ s12) {
  int tid = blockIdx.x * 256 + threadIdx.x;          // 0 .. 131071
  int n = tid >> 8;                                  // 0..511
  int k = tid & 255;                                 // 0..255
  Wt[n * GFIN + k] = (_Float16)W[k * GFOUT + n];
  s12[tid] = 0.0f;
}

// ---------------------------------------------------------------------------
// k1: ht = h@W + b via WMMA f16.  Each wave owns one 16x16 output tile.
//     Writes htT (f16, [b][h][d][j], j contiguous) and s1/s2 = ht.a1 / ht.a2
// ---------------------------------------------------------------------------
__global__ __launch_bounds__(256) void k1_gemm(const float* __restrict__ hin,
                                               const float* __restrict__ bias_g,
                                               const float* __restrict__ avec,
                                               const _Float16* __restrict__ Wt,
                                               _Float16* __restrict__ htT,
                                               float* __restrict__ s1,
                                               float* __restrict__ s2) {
  const int lane = threadIdx.x & 31;
  const int wave = threadIdx.x >> 5;
  const int tile = blockIdx.x * 8 + wave;    // 0 .. 16383
  const int mtile = tile >> 5;               // 0 .. 511  (rows of 16 over B*N)
  const int ntile = tile & 31;               // 0 .. 31   (cols of 16 over FOUT)
  const int m  = lane & 15;
  const int hi = lane >> 4;
  const int kbase = hi * 8;                  // K sub-range per lane half (ISA A/B layout)

  const float*    hrow = hin + (size_t)(mtile * 16 + m) * GFIN;
  const _Float16* wrow = Wt  + (size_t)(ntile * 16 + m) * GFIN;

  v8f c = {};
#pragma unroll
  for (int kc = 0; kc < 8; ++kc) {
    const int k0 = kc * 32;
    // A fragment: lane holds M=m, K = kbase+0..7 and 16+kbase+0..7
    v4f f0 = *(const v4f*)(hrow + k0 + kbase);
    v4f f1 = *(const v4f*)(hrow + k0 + kbase + 4);
    v4f f2 = *(const v4f*)(hrow + k0 + 16 + kbase);
    v4f f3 = *(const v4f*)(hrow + k0 + 16 + kbase + 4);
    v16h af;
#pragma unroll
    for (int i = 0; i < 4; ++i) {
      af[i]      = (_Float16)f0[i];
      af[4 + i]  = (_Float16)f1[i];
      af[8 + i]  = (_Float16)f2[i];
      af[12 + i] = (_Float16)f3[i];
    }
    // B fragment from Wt[n][k] (k contiguous): same per-lane K layout
    v8h blo = *(const v8h*)(wrow + k0 + kbase);
    v8h bhi = *(const v8h*)(wrow + k0 + 16 + kbase);
    v16h bf;
#pragma unroll
    for (int i = 0; i < 8; ++i) { bf[i] = blo[i]; bf[8 + i] = bhi[i]; }
    c = __builtin_amdgcn_wmma_f32_16x16x32_f16(false, af, false, bf,
                                               (short)0, c, false, false);
  }

  const int dg = ntile * 16 + m;             // global FOUT column
  const float bv = bias_g[dg];
#pragma unroll
  for (int r = 0; r < 8; ++r) c[r] += bv;

  const int head = dg >> 6;
  const int dl   = dg & 63;
  const float a1v = avec[dl];
  const float a2v = avec[GD + dl];
  const int bidx  = mtile >> 6;                      // batch
  const int jbase = (mtile & 63) * 16 + 8 * hi;      // node index of c[0]

  // s1/s2 partial dot over the 16 columns of this tile (reduce across lanes)
#pragma unroll
  for (int r = 0; r < 8; ++r) {
    float p1 = c[r] * a1v;
    float p2 = c[r] * a2v;
#pragma unroll
    for (int msk = 1; msk < 16; msk <<= 1) {
      p1 += __shfl_xor(p1, msk, 32);
      p2 += __shfl_xor(p2, msk, 32);
    }
    if (m == 0) {
      const int n = jbase + r;
      atomicAdd(&s1[((size_t)bidx * GN + n) * GH + head], p1);
      atomicAdd(&s2[((size_t)bidx * GN + n) * GH + head], p2);
    }
  }

  // transposed f16 store: htT[b][h][dl][j], 8 contiguous j per lane -> b128
  v8h hv;
#pragma unroll
  for (int r = 0; r < 8; ++r) hv[r] = (_Float16)c[r];
  *(v8h*)(htT + ((size_t)(bidx * GH + head) * GD + dl) * GN + jbase) = hv;
}

// ---------------------------------------------------------------------------
// k2: fused masked-softmax attention + attn@ht.
//     One wave per (batch, 16-row i-tile, head); online softmax over j.
// ---------------------------------------------------------------------------
__global__ __launch_bounds__(256) void k2_attn(const int* __restrict__ adj,
                                               const float* __restrict__ dist,
                                               const float* __restrict__ s1g,
                                               const float* __restrict__ s2g,
                                               const _Float16* __restrict__ htT,
                                               const float* __restrict__ dwp,
                                               const float* __restrict__ dbp,
                                               float* __restrict__ out) {
  __shared__ float s2sh[GH * GN];            // 32 KB: per-head s2 row
  const int lane  = threadIdx.x & 31;
  const int head  = threadIdx.x >> 5;        // wave id == head
  const int bidx  = blockIdx.x >> 6;
  const int itile = blockIdx.x & 63;
  const int m  = lane & 15;
  const int hi = lane >> 4;
  const int kbase = hi * 8;

  // stage s2[b, :, head] into LDS (stride-8 gather done once)
  for (int t = lane; t < GN; t += 32)
    s2sh[head * GN + t] = s2g[((size_t)bidx * GN + t) * GH + head];
  __syncthreads();
  const float* s2l = s2sh + head * GN;

  const int i = itile * 16 + m;
  const float s1v = s1g[((size_t)bidx * GN + i) * GH + head];
  const float dw = dwp[0];
  const float db = dbp[0];
  const int*   adjrow  = adj  + ((size_t)bidx * GN + i) * GN;
  const float* distrow = dist + ((size_t)bidx * GN + i) * GN;
  const _Float16* bbase = htT + (size_t)(bidx * GH + head) * GD * GN;

  float rowM = -1e30f;
  float rowS = 0.0f;
  v8f acc[4] = {};

  for (int jc = 0; jc < 32; ++jc) {
    const int j0 = jc * 32;
    // adjacency + distance for this lane's 16 (i,j) pairs (two 8-wide runs)
    v4i av0 = *(const v4i*)(adjrow + j0 + kbase);
    v4i av1 = *(const v4i*)(adjrow + j0 + kbase + 4);
    v4i av2 = *(const v4i*)(adjrow + j0 + 16 + kbase);
    v4i av3 = *(const v4i*)(adjrow + j0 + 16 + kbase + 4);
    v4f dv0 = *(const v4f*)(distrow + j0 + kbase);
    v4f dv1 = *(const v4f*)(distrow + j0 + kbase + 4);
    v4f dv2 = *(const v4f*)(distrow + j0 + 16 + kbase);
    v4f dv3 = *(const v4f*)(distrow + j0 + 16 + kbase + 4);
    if (jc < 31) {   // pull next chunk's lines toward L0/L2
      __builtin_prefetch(adjrow + j0 + 32 + kbase);
      __builtin_prefetch(distrow + j0 + 32 + kbase);
    }

    // masked logits, laid out exactly in the 16-bit A-fragment K order
    float e[16];
#pragma unroll
    for (int t = 0; t < 16; ++t) {
      const int jj = j0 + ((t < 8) ? (kbase + t) : (16 + kbase + (t - 8)));
      const int   ad = (t < 4) ? av0[t] : (t < 8) ? av1[t - 4]
                     : (t < 12) ? av2[t - 8] : av3[t - 12];
      const float ds = (t < 4) ? dv0[t] : (t < 8) ? dv1[t - 4]
                     : (t < 12) ? dv2[t - 8] : dv3[t - 12];
      const float su = s1v + s2l[jj];
      const float lr = fmaxf(su, 0.0f) + GALPHA * fminf(su, 0.0f);
      const float ev = lr + dw * __builtin_amdgcn_rcpf(ds + 1e-8f) + db;
      e[t] = (ad == 0) ? -1.0e9f : ev;
    }

    // online softmax: row stats live in registers, lane pair (l, l+16) = one row
    float mc = e[0];
#pragma unroll
    for (int t = 1; t < 16; ++t) mc = fmaxf(mc, e[t]);
    mc = fmaxf(mc, __shfl_xor(mc, 16, 32));
    const float newM  = fmaxf(rowM, mc);
    const float scale = __expf(rowM - newM);
    rowM = newM;

    float ps = 0.0f;
    v16h af;
#pragma unroll
    for (int t = 0; t < 16; ++t) {
      const float p = __expf(e[t] - newM);
      ps += p;
      af[t] = (_Float16)p;
    }
    ps += __shfl_xor(ps, 16, 32);
    rowS = rowS * scale + ps;

    // rescale accumulators: row r+8*hi needs scale held by lane (8*hi + r)
    float sc[8];
#pragma unroll
    for (int r = 0; r < 8; ++r) sc[r] = __shfl(scale, 8 * hi + r, 32);
#pragma unroll
    for (int tt = 0; tt < 4; ++tt)
#pragma unroll
      for (int r = 0; r < 8; ++r) acc[tt][r] *= sc[r];

    // 4 d-tiles of 16: B fragments are two contiguous b128 loads from htT
#pragma unroll
    for (int tt = 0; tt < 4; ++tt) {
      const _Float16* bp = bbase + (size_t)(tt * 16 + m) * GN + j0;
      v8h blo = *(const v8h*)(bp + kbase);
      v8h bhi = *(const v8h*)(bp + 16 + kbase);
      v16h bf;
#pragma unroll
      for (int q = 0; q < 8; ++q) { bf[q] = blo[q]; bf[8 + q] = bhi[q]; }
      acc[tt] = __builtin_amdgcn_wmma_f32_16x16x32_f16(false, af, false, bf,
                                                       (short)0, acc[tt],
                                                       false, false);
    }
  }

  // epilogue: divide by row sums and store fp32
  float inv[8];
#pragma unroll
  for (int r = 0; r < 8; ++r) {
    const float s = __shfl(rowS, 8 * hi + r, 32);
    inv[r] = __builtin_amdgcn_rcpf(s);
  }
#pragma unroll
  for (int tt = 0; tt < 4; ++tt)
#pragma unroll
    for (int r = 0; r < 8; ++r) {
      const int row = itile * 16 + 8 * hi + r;
      out[((size_t)bidx * GN + row) * GFOUT + head * GD + tt * 16 + m] =
          acc[tt][r] * inv[r];
    }
}

// ---------------------------------------------------------------------------
extern "C" void kernel_launch(void* const* d_in, const int* in_sizes, int n_in,
                              void* d_out, int out_size, void* d_ws, size_t ws_size,
                              hipStream_t stream) {
  const float* h    = (const float*)d_in[0];
  const int*   adj  = (const int*)d_in[1];
  const float* dist = (const float*)d_in[2];
  const float* W    = (const float*)d_in[3];
  const float* bvec = (const float*)d_in[4];
  const float* avec = (const float*)d_in[5];
  const float* dwp  = (const float*)d_in[6];
  const float* dbp  = (const float*)d_in[7];
  float* out = (float*)d_out;

  // workspace layout
  char* ws = (char*)d_ws;
  _Float16* Wt  = (_Float16*)ws;                         // 512*256 f16 = 256 KB
  _Float16* htT = (_Float16*)(ws + 524288);              // 8*8*64*1024 f16 = 8 MB
  float*    s1  = (float*)(ws + 524288 + 8388608);       // 65536 f32
  float*    s2  = s1 + GB * GN * GH;                     // 65536 f32

  k0_prep<<<512, 256, 0, stream>>>(W, Wt, s1);           // also zeros s1|s2
  k1_gemm<<<2048, 256, 0, stream>>>(h, bvec, avec, Wt, htT, s1, s2);
  k2_attn<<<512, 256, 0, stream>>>(adj, dist, s1, s2, htT, dwp, dbp, out);
}
